// SimpleSFModel_53369263620272
// MI455X (gfx1250) — matrix operand order repaired and checked by reading, not compiled, not run on hardware
//
#include <hip/hip_runtime.h>
#include <math.h>

typedef __attribute__((ext_vector_type(16))) _Float16 v16h;
typedef __attribute__((ext_vector_type(8)))  float    v8f;

#define NB 2048
#define CUT 10
#define PLANE 10240                      // f32 plane stride (>=10000)
#define CX_RE 0
#define CX_IM (1*PLANE)
#define BS_RE (2*PLANE)
#define BS_IM (3*PLANE)
#define M_OFF (4*PLANE)                  // 8 matrices * 200 floats
#define V_OFF (M_OFF + 4096)             // 2048*4*10*2 = 163840 floats
#define SCR_OFF (V_OFF + 163840)         // 2 blocks * 6 * PLANE
#define GATE_OFF_BYTES ((size_t)(SCR_OFF + 2*6*PLANE) * 4)
#define GPLANE 14336                     // 112*128 halves per plane
#define GSLOT  (3*GPLANE)                // re, im, -im planes per gate

// ---------------- small complex 10x10 helpers (per-thread) ----------------
__device__ void cmm10(const float* Ar, const float* Ai, const float* Br, const float* Bi,
                      float* Cr, float* Ci) {
  for (int i = 0; i < 10; ++i)
    for (int j = 0; j < 10; ++j) {
      float sr = 0.f, si = 0.f;
      for (int k = 0; k < 10; ++k) {
        float ar = Ar[i*10+k], ai = Ai[i*10+k];
        float br = Br[k*10+j], bi = Bi[k*10+j];
        sr += ar*br - ai*bi;
        si += ar*bi + ai*br;
      }
      Cr[i*10+j] = sr; Ci[i*10+j] = si;
    }
}

// expm via scaling-and-squaring + order-12 Taylor (Horner). X is destroyed.
__device__ void cexpm10(float* Xr, float* Xi, float* Er, float* Ei) {
  float nrm = 0.f;
  for (int j = 0; j < 10; ++j) {
    float s = 0.f;
    for (int i = 0; i < 10; ++i) s += fabsf(Xr[i*10+j]) + fabsf(Xi[i*10+j]);
    nrm = fmaxf(nrm, s);
  }
  int sc = 0;
  while (nrm > 0.5f && sc < 30) { nrm *= 0.5f; ++sc; }
  float f = exp2f(-(float)sc);
  for (int e = 0; e < 100; ++e) { Xr[e] *= f; Xi[e] *= f; }
  float Tr[100], Ti[100], Pr[100], Pi[100];
  const float inv12 = 1.f / 12.f;
  for (int e = 0; e < 100; ++e) { Tr[e] = Xr[e]*inv12; Ti[e] = Xi[e]*inv12; }
  for (int d = 0; d < 10; ++d) Tr[d*11] += 1.f;
  for (int k = 11; k >= 1; --k) {
    cmm10(Xr, Xi, Tr, Ti, Pr, Pi);
    float ik = 1.f/(float)k;
    for (int e = 0; e < 100; ++e) { Tr[e] = Pr[e]*ik; Ti[e] = Pi[e]*ik; }
    for (int d = 0; d < 10; ++d) Tr[d*11] += 1.f;
  }
  for (int s2 = 0; s2 < sc; ++s2) {
    cmm10(Tr, Ti, Tr, Ti, Pr, Pi);
    for (int e = 0; e < 100; ++e) { Tr[e] = Pr[e]; Ti[e] = Pi[e]; }
  }
  for (int e = 0; e < 100; ++e) { Er[e] = Tr[e]; Ei[e] = Ti[e]; }
}

__device__ void build_squeeze_gen(float zr, float zi, float* Xr, float* Xi) {
  for (int e = 0; e < 100; ++e) { Xr[e] = 0.f; Xi[e] = 0.f; }
  for (int i = 0; i + 2 < 10; ++i) {
    float c = 0.5f * sqrtf((float)((i+1)*(i+2)));
    Xr[i*10 + i + 2]  =  zr*c;  Xi[i*10 + i + 2]  = -zi*c;  // 0.5*conj(z)*a^2
    Xr[(i+2)*10 + i]  = -zr*c;  Xi[(i+2)*10 + i]  = -zi*c;  // -0.5*z*(a^+)^2
  }
}

__device__ void build_disp_gen(float ar, float ai, float* Xr, float* Xi) {
  for (int e = 0; e < 100; ++e) { Xr[e] = 0.f; Xi[e] = 0.f; }
  for (int i = 1; i < 10; ++i) {            // alpha * a^+ : entry (i, i-1) = alpha*sqrt(i)
    float c = sqrtf((float)i);
    Xr[i*10 + i - 1] = ar*c;  Xi[i*10 + i - 1] = ai*c;
  }
  for (int i = 0; i + 1 < 10; ++i) {        // -conj(alpha)*a : entry (i, i+1)
    float c = sqrtf((float)(i+1));
    Xr[i*10 + i + 1] = -ar*c; Xi[i*10 + i + 1] = ai*c;
  }
}

// ---------------- kernel 1: per-(b,w) encoding vector v = D_enc @ S_enc |0> --------
__global__ void encode_kernel(const float* __restrict__ jets,
                              const float* __restrict__ s_scale_p,
                              float* __restrict__ ws) {
  int t = blockIdx.x * blockDim.x + threadIdx.x;
  if (t >= NB * 4) return;
  float eta = jets[t*3 + 0], ph = jets[t*3 + 1], pt = jets[t*3 + 2];
  float s_scale = s_scale_p[0];
  float scale = 10.f / (1.f + expf(-s_scale)) + 0.01f;

  float Xr[100], Xi[100], Sr[100], Si[100], Dr[100], Di[100];
  float sphi = pt * ph * 0.5f;
  build_squeeze_gen(eta * cosf(sphi), eta * sinf(sphi), Xr, Xi);
  cexpm10(Xr, Xi, Sr, Si);
  float mag = scale * pt;
  build_disp_gen(mag * cosf(eta), mag * sinf(eta), Xr, Xi);
  cexpm10(Xr, Xi, Dr, Di);

  float* V = ws + V_OFF + (size_t)t * 20;
  for (int i = 0; i < 10; ++i) {          // v = D @ S[:,0]
    float vr = 0.f, vi = 0.f;
    for (int k = 0; k < 10; ++k) {
      float br = Sr[k*10], bi = Si[k*10];
      vr += Dr[i*10+k]*br - Di[i*10+k]*bi;
      vi += Dr[i*10+k]*bi + Di[i*10+k]*br;
    }
    V[2*i] = vr; V[2*i+1] = vi;
  }
}

// ---------------- kernel 2: per-layer per-wire M = D_lay @ S_lay -------------------
__global__ void layer_kernel(const float* __restrict__ disp_mag,
                             const float* __restrict__ disp_phase,
                             const float* __restrict__ squeeze_mag,
                             const float* __restrict__ squeeze_phase,
                             float* __restrict__ ws) {
  int t = threadIdx.x;
  if (t >= 8) return;                      // t = L*4 + w
  float Xr[100], Xi[100], Sr[100], Si[100], Dr[100], Di[100], Mr[100], Mi[100];
  float r = squeeze_mag[t], p = squeeze_phase[t];
  build_squeeze_gen(r*cosf(p), r*sinf(p), Xr, Xi);
  cexpm10(Xr, Xi, Sr, Si);
  float dm = disp_mag[t], dp = disp_phase[t];
  build_disp_gen(dm*cosf(dp), dm*sinf(dp), Xr, Xi);
  cexpm10(Xr, Xi, Dr, Di);
  cmm10(Dr, Di, Sr, Si, Mr, Mi);
  float* out = ws + M_OFF + (size_t)t * 200;
  for (int e = 0; e < 100; ++e) { out[e] = Mr[e]; out[100+e] = Mi[e]; }
}

// ---------------- kernel 3: 100x100 expm for CX (blk 0) and BS (blk 1) -------------
__device__ inline void gsync() { __threadfence(); __syncthreads(); }

__global__ void bigexp_kernel(float* __restrict__ ws) {
  const int which = blockIdx.x;            // 0 = CX, 1 = BS
  const int tid = threadIdx.x;
  float* Xr = ws + SCR_OFF + (size_t)which * 6 * PLANE;
  float* Xi = Xr + PLANE;
  float* Tr = Xi + PLANE;  float* Ti = Tr + PLANE;
  float* Pr = Ti + PLANE;  float* Pi = Pr + PLANE;
  float* Or = ws + (which ? BS_RE : CX_RE);
  float* Oi = ws + (which ? BS_IM : CX_IM);

  for (int e = tid; e < 10000; e += 256) {   // build generator, pre-scaled by 2^-10
    int pp = e / 100, q = e % 100;
    int i = pp / 10, j = pp % 10, k = q / 10, l = q % 10;
    float gr = 0.f, gi = 0.f;
    if (which == 0) {                        // CX: -1/2 * (a+ad)[i,k] * (a-ad)[j,l]
      float A = (k == i+1) ? sqrtf((float)k) : ((k == i-1) ? sqrtf((float)i) : 0.f);
      float Bm = (l == j+1) ? sqrtf((float)l) : ((l == j-1) ? -sqrtf((float)j) : 0.f);
      gr = -0.5f * A * Bm;
    } else {                                 // BS: i*(pi/4)*(a_ik*ad_jl + ad_ik*a_jl)
      float t1 = ((k == i+1) ? sqrtf((float)k) : 0.f) * ((l == j-1) ? sqrtf((float)j) : 0.f);
      float t2 = ((k == i-1) ? sqrtf((float)i) : 0.f) * ((l == j+1) ? sqrtf((float)l) : 0.f);
      gi = 0.78539816339f * (t1 + t2);
    }
    const float f = 1.f / 1024.f;
    Xr[e] = gr * f; Xi[e] = gi * f;
  }
  gsync();
  for (int e = tid; e < 10000; e += 256) {   // T = I + X/12
    float t12 = 1.f/12.f;
    Tr[e] = Xr[e]*t12; Ti[e] = Xi[e]*t12;
    if (e/100 == e%100) Tr[e] += 1.f;
  }
  gsync();
  for (int kk = 11; kk >= 1; --kk) {
    for (int e = tid; e < 10000; e += 256) { // P = X*T
      int row = e / 100, col = e % 100;
      float sr = 0.f, si = 0.f;
      for (int m = 0; m < 100; ++m) {
        float ar = Xr[row*100+m], ai = Xi[row*100+m];
        float br = Tr[m*100+col], bi = Ti[m*100+col];
        sr += ar*br - ai*bi; si += ar*bi + ai*br;
      }
      Pr[e] = sr; Pi[e] = si;
    }
    gsync();
    float ik = 1.f/(float)kk;
    for (int e = tid; e < 10000; e += 256) {
      Tr[e] = Pr[e]*ik; Ti[e] = Pi[e]*ik;
      if (e/100 == e%100) Tr[e] += 1.f;
    }
    gsync();
  }
  for (int sq = 0; sq < 10; ++sq) {          // T = T*T, ten times
    for (int e = tid; e < 10000; e += 256) {
      int row = e / 100, col = e % 100;
      float sr = 0.f, si = 0.f;
      for (int m = 0; m < 100; ++m) {
        float ar = Tr[row*100+m], ai = Ti[row*100+m];
        float br = Tr[m*100+col], bi = Ti[m*100+col];
        sr += ar*br - ai*bi; si += ar*bi + ai*br;
      }
      Pr[e] = sr; Pi[e] = si;
    }
    gsync();
    for (int e = tid; e < 10000; e += 256) { Tr[e] = Pr[e]; Ti[e] = Pi[e]; }
    gsync();
  }
  for (int e = tid; e < 10000; e += 256) { Or[e] = Tr[e]; Oi[e] = Ti[e]; }
}

// ---------------- kernel 4: fold layer M's into BS copies, emit padded f16 planes --
// slots: 0=CX 1=BS01 2=BS12_L0 3=BS23_L0 4=BS30_L0 5=BS12_L1 6=BS23_L1 7=BS30_L1
__global__ void fold_kernel(const float* __restrict__ ws, _Float16* __restrict__ g16) {
  const int blk = blockIdx.x, tid = threadIdx.x;
  const float* BSr = ws + BS_RE; const float* BSi = ws + BS_IM;
  const float* CXr = ws + CX_RE; const float* CXi = ws + CX_IM;
  _Float16* base = g16 + (size_t)blk * GSLOT;

  for (int e = tid; e < 112*128; e += 256) {
    int p = e / 128, k = e % 128;
    float fr = 0.f, fi = 0.f;
    if (p < 100 && k < 100) {
      if (blk == 0)      { fr = CXr[p*100+k]; fi = CXi[p*100+k]; }
      else if (blk == 1) { fr = BSr[p*100+k]; fi = BSi[p*100+k]; }
      else {
        int L = (blk - 2) / 3, which = (blk - 2) % 3;
        int i = p / 10, j = p % 10;
        if (which < 2) {                   // fold M[wire] on first pair index
          int wf = (which == 0) ? 1 : 2;
          const float* Mr = ws + M_OFF + (size_t)(L*4 + wf) * 200;
          const float* Mi = Mr + 100;
          for (int ip = 0; ip < 10; ++ip) {
            float mr = Mr[i*10+ip], mi = Mi[i*10+ip];
            float br = BSr[(ip*10+j)*100+k], bi = BSi[(ip*10+j)*100+k];
            fr += mr*br - mi*bi; fi += mr*bi + mi*br;
          }
        } else {                           // BS(3,0): fold M3 on i, M0 on j
          const float* M3r = ws + M_OFF + (size_t)(L*4 + 3) * 200; const float* M3i = M3r + 100;
          const float* M0r = ws + M_OFF + (size_t)(L*4 + 0) * 200; const float* M0i = M0r + 100;
          for (int ip = 0; ip < 10; ++ip) {
            float ir = 0.f, ii = 0.f;
            for (int jp = 0; jp < 10; ++jp) {
              float mr = M0r[j*10+jp], mi = M0i[j*10+jp];
              float br = BSr[(ip*10+jp)*100+k], bi = BSi[(ip*10+jp)*100+k];
              ir += mr*br - mi*bi; ii += mr*bi + mi*br;
            }
            float mr = M3r[i*10+ip], mi = M3i[i*10+ip];
            fr += mr*ir - mi*ii; fi += mr*ii + mi*ir;
          }
        }
      }
    }
    base[e]            = (_Float16)fr;
    base[GPLANE + e]   = (_Float16)fi;
    base[2*GPLANE + e] = (_Float16)(-fi);   // pre-negated imag for the "Cre" chain
  }
}

// ---------------- WMMA fragment loaders (gfx1250 wave32 layouts) -------------------
__device__ inline v16h load_afrag(const _Float16* __restrict__ plane,
                                  int mt, int kt, int lane) {
  int M = mt*16 + (lane & 15);
  bool hi = lane >= 16;
  const _Float16* row = plane + M * 128;
  int kbase = kt*32 + (hi ? 8 : 0);
  union { v16h v; _Float16 h[16]; } u;
#pragma unroll
  for (int j = 0; j < 8; ++j) {
    int k = kbase + 2*(j & 3) + ((j >= 4) ? 16 : 0);
    u.h[2*j]   = row[k];
    u.h[2*j+1] = row[k+1];
  }
  return u.v;
}

__device__ inline void load_bfrag(const _Float16* __restrict__ Sr,
                                  const _Float16* __restrict__ Si,
                                  int nt, int kt, int lane,
                                  int sm1, int sm2, int qoff,
                                  v16h& br, v16h& bi) {
  int kbase = kt*32 + ((lane >= 16) ? 8 : 0);
  union { v16h v; _Float16 h[16]; } ur, ui;
#pragma unroll
  for (int j = 0; j < 8; ++j) {
    int k0 = kbase + 2*(j & 3) + ((j >= 4) ? 16 : 0);
    int k1 = k0 + 1;
    k0 = (k0 < 100) ? k0 : 99;   // clamped reads; k>=100 killed by zero-padded A
    k1 = (k1 < 100) ? k1 : 99;
    int a0 = (k0/10)*sm1 + (k0%10)*sm2 + qoff;
    int a1 = (k1/10)*sm1 + (k1%10)*sm2 + qoff;
    ur.h[2*j] = Sr[a0]; ur.h[2*j+1] = Sr[a1];
    ui.h[2*j] = Si[a0]; ui.h[2*j+1] = Si[a1];
  }
  br = ur.v; bi = ui.v;
}

// ---------------- kernel 5: whole circuit per batch element in LDS -----------------
__global__ __launch_bounds__(256) void cvqnn_main(const float* __restrict__ ws,
                                                  const _Float16* __restrict__ g16,
                                                  const float* __restrict__ wdense,
                                                  const float* __restrict__ bdense,
                                                  float* __restrict__ out) {
  extern __shared__ _Float16 smem[];        // 40000 halves = 80 KB
  _Float16* s0r = smem;
  _Float16* s0i = smem + 10000;
  _Float16* s1r = smem + 20000;
  _Float16* s1i = smem + 30000;
  __shared__ float vr_s[40], vi_s[40];
  __shared__ float red[256];

  const int b = blockIdx.x;
  const int tid = threadIdx.x;
  const int lane = tid & 31;
  const int wave = tid >> 5;

  // ---- build product state from encoding vectors ----
  if (tid < 40) {
    const float* V = ws + V_OFF + (size_t)b * 80;
    vr_s[tid] = V[2*tid];
    vi_s[tid] = V[2*tid + 1];
  }
  __syncthreads();
  for (int e = tid; e < 10000; e += 256) {
    int i0 = e / 1000, r1 = e % 1000;
    int i1 = r1 / 100, r2 = r1 % 100;
    int i2 = r2 / 10,  i3 = r2 % 10;
    float ar = vr_s[i0],      ai = vi_s[i0];
    float br = vr_s[10 + i1], bi = vi_s[10 + i1];
    float tr = ar*br - ai*bi, ti = ar*bi + ai*br;
    br = vr_s[20 + i2]; bi = vi_s[20 + i2];
    ar = tr*br - ti*bi; ai = tr*bi + ti*br;
    br = vr_s[30 + i3]; bi = vi_s[30 + i3];
    tr = ar*br - ai*bi; ti = ar*bi + ai*br;
    s0r[e] = (_Float16)tr; s0i[e] = (_Float16)ti;
  }
  __syncthreads();

  // ---- gate sequence: 2 layers x (6 CX + BS01 + 3 folded BS) ----
  const int gslot[20] = {0,0,0,0,0,0, 1,2,3,4, 0,0,0,0,0,0, 1,5,6,7};
  const int gm1[20]   = {0,0,0,1,1,2, 0,1,2,3, 0,0,0,1,1,2, 0,1,2,3};
  const int gm2[20]   = {1,2,3,2,3,3, 1,2,3,0, 1,2,3,2,3,3, 1,2,3,0};
  const int str[4] = {1000, 100, 10, 1};

  for (int g = 0; g < 20; ++g) {
    const int m1 = gm1[g], m2 = gm2[g];
    const int sm1 = str[m1], sm2 = str[m2];
    int oa = -1, ob = -1;
    for (int w = 0; w < 4; ++w)
      if (w != m1 && w != m2) { if (oa < 0) oa = w; else ob = w; }
    const int sa = str[oa], sb = str[ob];

    const _Float16* Sre = (g & 1) ? s1r : s0r;
    const _Float16* Sim = (g & 1) ? s1i : s0i;
    _Float16* Dre = (g & 1) ? s0r : s1r;
    _Float16* Dim = (g & 1) ? s0i : s1i;

    const _Float16* Gr = g16 + (size_t)gslot[g] * GSLOT;
    const _Float16* Gi = Gr + GPLANE;
    const _Float16* Gn = Gi + GPLANE;      // -imag
    __builtin_prefetch((const void*)Gr, 0, 1);

    const int N = (lane & 15);
    const bool hi = lane >= 16;

    for (int t = wave; t < 49; t += 8) {
      int mt = t / 7, nt = t % 7;
      int Nq = nt*16 + N;
      int qc = (Nq < 100) ? Nq : 99;       // clamp; cols >= 100 discarded at store
      int qoff = (qc/10)*sa + (qc%10)*sb;

      // 4 independent accumulator chains -> one WMMA per chain per K-chunk,
      // so dependent WMMAs are >=3 ops apart (hazard window hidden by work).
      v8f cr0 = {}; v8f cr1 = {}; v8f ci0 = {}; v8f ci1 = {};
#pragma unroll
      for (int kt = 0; kt < 4; ++kt) {
        v16h ar = load_afrag(Gr, mt, kt, lane);
        v16h aiF = load_afrag(Gi, mt, kt, lane);
        v16h an = load_afrag(Gn, mt, kt, lane);
        v16h brF, biF;
        load_bfrag(Sre, Sim, nt, kt, lane, sm1, sm2, qoff, brF, biF);
        // Cre = Ar*Br + (-Ai)*Bi ; Cim = Ar*Bi + Ai*Br  (split chains)
        cr0 = __builtin_amdgcn_wmma_f32_16x16x32_f16(false, ar,  false, brF, (short)0, cr0, false, false);
        cr1 = __builtin_amdgcn_wmma_f32_16x16x32_f16(false, an,  false, biF, (short)0, cr1, false, false);
        ci0 = __builtin_amdgcn_wmma_f32_16x16x32_f16(false, ar,  false, biF, (short)0, ci0, false, false);
        ci1 = __builtin_amdgcn_wmma_f32_16x16x32_f16(false, aiF, false, brF, (short)0, ci1, false, false);
      }

      if (Nq < 100) {
#pragma unroll
        for (int r = 0; r < 8; ++r) {
          int M = mt*16 + r + (hi ? 8 : 0);
          if (M < 100) {
            int addr = (M/10)*sm1 + (M%10)*sm2 + qoff;
            Dre[addr] = (_Float16)(cr0[r] + cr1[r]);
            Dim[addr] = (_Float16)(ci0[r] + ci1[r]);
          }
        }
      }
    }
    __syncthreads();
  }

  // ---- readout: sum |amp|^2 * (i0*w0 + i1*w1 + i2*w2), + bias ----
  const float w0 = wdense[0], w1 = wdense[1], w2 = wdense[2];
  float acc = 0.f;
  for (int e = tid; e < 10000; e += 256) {
    float re = (float)s0r[e], im = (float)s0i[e];
    int i0 = e / 1000, i1 = (e / 100) % 10, i2 = (e / 10) % 10;
    acc += (re*re + im*im) * ((float)i0*w0 + (float)i1*w1 + (float)i2*w2);
  }
  red[tid] = acc;
  __syncthreads();
  for (int off = 128; off > 0; off >>= 1) {
    if (tid < off) red[tid] += red[tid + off];
    __syncthreads();
  }
  if (tid == 0) out[b] = red[0] + bdense[0];
}

// ---------------- host launcher ----------------------------------------------------
extern "C" void kernel_launch(void* const* d_in, const int* in_sizes, int n_in,
                              void* d_out, int out_size, void* d_ws, size_t ws_size,
                              hipStream_t stream) {
  const float* jets          = (const float*)d_in[0];
  const float* s_scale       = (const float*)d_in[1];
  const float* disp_mag      = (const float*)d_in[2];
  const float* disp_phase    = (const float*)d_in[3];
  const float* squeeze_mag   = (const float*)d_in[4];
  const float* squeeze_phase = (const float*)d_in[5];
  const float* w_dense       = (const float*)d_in[6];
  const float* b_dense       = (const float*)d_in[7];
  float* out = (float*)d_out;
  float* wsf = (float*)d_ws;
  _Float16* g16 = (_Float16*)((char*)d_ws + GATE_OFF_BYTES);

  encode_kernel<<<(NB*4 + 127)/128, 128, 0, stream>>>(jets, s_scale, wsf);
  layer_kernel<<<1, 8, 0, stream>>>(disp_mag, disp_phase, squeeze_mag, squeeze_phase, wsf);
  bigexp_kernel<<<2, 256, 0, stream>>>(wsf);
  fold_kernel<<<8, 256, 0, stream>>>(wsf, g16);
  cvqnn_main<<<NB, 256, 80000, stream>>>(wsf, g16, w_dense, b_dense, out);
}
